// AttnBlock_12704513262242
// MI455X (gfx1250) — matrix operand order repaired and checked by reading, not compiled, not run on hardware
//
#include <hip/hip_runtime.h>

#define B_ 32
#define W_ 2048
#define NT 64            // N tile per block
#define KB 32            // K per WMMA step
#define KSPLIT 4
#define KLEN (W_ / KSPLIT)
#define NIT (KLEN / KB)  // 16 K-steps per block

typedef __attribute__((ext_vector_type(16))) __bf16 v16bf;
typedef __attribute__((ext_vector_type(8)))  __bf16 v8bf;
typedef __attribute__((ext_vector_type(8)))  float  v8f;

// ---------------- LayerNorm: one block (256 thr) per row ----------------
__global__ void ln_kernel(const float* __restrict__ x, const float* __restrict__ gamma,
                          const float* __restrict__ beta, __bf16* __restrict__ h) {
    __shared__ float red[256];
    const int row = blockIdx.x;
    const int tid = threadIdx.x;
    const float* xr = x + (size_t)row * W_;
    float vals[8];
    float s = 0.f;
    #pragma unroll
    for (int i = 0; i < 8; ++i) { vals[i] = xr[tid + i * 256]; s += vals[i]; }
    red[tid] = s; __syncthreads();
    for (int off = 128; off > 0; off >>= 1) {
        if (tid < off) red[tid] += red[tid + off];
        __syncthreads();
    }
    const float mu = red[0] * (1.f / W_);
    __syncthreads();
    float v2 = 0.f;
    #pragma unroll
    for (int i = 0; i < 8; ++i) { float d = vals[i] - mu; v2 = fmaf(d, d, v2); }
    red[tid] = v2; __syncthreads();
    for (int off = 128; off > 0; off >>= 1) {
        if (tid < off) red[tid] += red[tid + off];
        __syncthreads();
    }
    const float rstd = rsqrtf(red[0] * (1.f / W_) + 1e-6f);
    #pragma unroll
    for (int i = 0; i < 8; ++i) {
        const int c = tid + i * 256;
        const float hn = (vals[i] - mu) * rstd * gamma[c] + beta[c];
        h[(size_t)row * W_ + c] = (__bf16)hn;
    }
}

// ------- Skinny GEMM: part[split] = A(bf16,32xW) @ Wm(f32,WxW) K-slice -------
// 8 waves/block, one 16x16 WMMA tile per wave (2 M-tiles x 4 N-tiles).
// Weights stream HBM -> LDS via async-LDS loads (ASYNCcnt), double-buffered.
__global__ void gemm_kernel(const __bf16* __restrict__ A, const float* __restrict__ Wm,
                            float* __restrict__ part) {
    __shared__ float tile[2][KB][NT + 4];   // two f32 staged tiles, padded pitch 68
    const int tid  = threadIdx.x;
    const int lane = tid & 31;
    const int wave = tid >> 5;
    const int nb   = blockIdx.x * NT;
    const int kbeg = blockIdx.y * KLEN;
    const int mt   = wave & 1;
    const int nt   = wave >> 1;
    // A frag: lane<16 -> M=lane, K={0..7,16..23}; lane>=16 -> M=lane-16, K={8..15,24..31}
    const int am   = mt * 16 + (lane & 15);
    const int klo  = (lane >> 4) * 8;
    // B frag: lane<16 -> N=lane, K={0..7,16..23}; lane>=16 -> N=lane-16, K={8..15,24..31}
    const int bn   = nt * 16 + (lane & 15);
    const int bks  = (lane >> 4) * 8;
    // cooperative W-tile staging: 8 threads per k-row, 8 floats (32B) each
    const int lk = tid >> 3;
    const int ln = (tid & 7) * 8;
    const unsigned lds0 = (unsigned)(size_t)&tile[0][lk][ln];
    const unsigned lds1 = (unsigned)(size_t)&tile[1][lk][ln];

    // issue async copy of one 32x64 f32 W-tile (this thread's 32B) into buffer
    auto issue = [&](int kb, int buf) {
        const float* g = Wm + (size_t)(kb + lk) * W_ + nb + ln;
        const unsigned l = buf ? lds1 : lds0;
        asm volatile("global_load_async_to_lds_b128 %0, %1, off"
                     :: "v"(l), "v"(g) : "memory");
        asm volatile("global_load_async_to_lds_b128 %0, %1, off offset:16"
                     :: "v"(l), "v"(g) : "memory");
    };

    v8f acc = {};
    auto step = [&](int kb, int buf) {
        union { v16bf v; v8bf h[2]; } a;
        const __bf16* ap = A + (size_t)am * W_ + kb + klo;
        a.h[0] = *(const v8bf*)(ap);
        a.h[1] = *(const v8bf*)(ap + 16);
        v16bf b;
        #pragma unroll
        for (int i = 0; i < 16; ++i) b[i] = (__bf16)tile[buf][bks + i + (i & 8)][bn];
        acc = __builtin_amdgcn_wmma_f32_16x16x32_bf16(
            false, a.v, false, b, (short)0, acc, false, false);
    };

    issue(kbeg, 0);
    asm volatile("s_wait_asynccnt 0" ::: "memory");
    __syncthreads();

    for (int i = 0; i < NIT; i += 2) {
        const int kb = kbeg + i * KB;
        if (i + 1 < NIT) issue(kb + KB, 1);       // prefetch into buf1
        step(kb, 0);                              // compute on buf0 (overlaps copy)
        asm volatile("s_wait_asynccnt 0" ::: "memory");
        __syncthreads();
        if (i + 2 < NIT) issue(kb + 2 * KB, 0);   // prefetch into buf0
        step(kb + KB, 1);                         // compute on buf1
        asm volatile("s_wait_asynccnt 0" ::: "memory");
        __syncthreads();
    }

    // C layout: lanes 0-15 hold rows r, lanes 16-31 rows r+8; N = lane&15
    const int mrow = mt * 16 + ((lane >> 4) << 3);
    const int ncol = nb + nt * 16 + (lane & 15);
    float* op = part + (size_t)blockIdx.y * (B_ * W_) + (size_t)mrow * W_ + ncol;
    #pragma unroll
    for (int r = 0; r < 8; ++r) op[(size_t)r * W_] = acc[r];
}

// ----- deterministic split-K reduction: out = bias + (resid?) + sum parts -----
__global__ void reduce_kernel(const float* __restrict__ part, const float* __restrict__ bias,
                              const float* __restrict__ resid, float* __restrict__ out) {
    const int idx = blockIdx.x * 256 + threadIdx.x;
    const int n = idx & (W_ - 1);
    float s = bias[n];
    if (resid) s += resid[idx];
    #pragma unroll
    for (int sp = 0; sp < KSPLIT; ++sp) s += part[(size_t)sp * (B_ * W_) + idx];
    out[idx] = s;
}

// --- softmin attention: h2[b,j] = sum_i exp(-|q_j-k_i|) v_i / sum_i exp(-|q_j-k_i|) ---
__global__ void softmin_kernel(const float* __restrict__ Q, const float* __restrict__ K,
                               const float* __restrict__ V, __bf16* __restrict__ h2) {
    __shared__ float2 kv[W_];                        // interleaved (k, v): 1 ds_load_b64/iter
    const int b = blockIdx.x >> 3;                   // 8 j-chunks per batch row
    const int j = ((blockIdx.x & 7) << 8) + threadIdx.x;
    const float* Kb = K + (size_t)b * W_;
    const float* Vb = V + (size_t)b * W_;
    for (int i = threadIdx.x; i < W_; i += 256) kv[i] = make_float2(Kb[i], Vb[i]);
    __syncthreads();
    const float qj = Q[(size_t)b * W_ + j];
    float num = 0.f, den = 0.f;
    #pragma unroll 8
    for (int i = 0; i < W_; ++i) {
        const float2 c = kv[i];
        const float e = __expf(-fabsf(qj - c.x));    // d>=0 -> e in (0,1], no overflow
        num = fmaf(e, c.y, num);
        den += e;
    }
    h2[(size_t)b * W_ + j] = (__bf16)(num / den);
}

extern "C" void kernel_launch(void* const* d_in, const int* in_sizes, int n_in,
                              void* d_out, int out_size, void* d_ws, size_t ws_size,
                              hipStream_t stream) {
    const float* x     = (const float*)d_in[0];
    const float* gamma = (const float*)d_in[1];
    const float* beta  = (const float*)d_in[2];
    const float* Wq    = (const float*)d_in[3];
    const float* bq    = (const float*)d_in[4];
    const float* Wk    = (const float*)d_in[5];
    const float* bk    = (const float*)d_in[6];
    const float* Wv    = (const float*)d_in[7];
    const float* bv    = (const float*)d_in[8];
    const float* Wp    = (const float*)d_in[9];
    const float* bp    = (const float*)d_in[10];
    float* out = (float*)d_out;

    const size_t NE = (size_t)B_ * W_;
    char* ws = (char*)d_ws;
    __bf16* h_bf  = (__bf16*)ws;  ws += NE * sizeof(__bf16);
    __bf16* h2_bf = (__bf16*)ws;  ws += NE * sizeof(__bf16);
    float*  q     = (float*)ws;   ws += NE * sizeof(float);
    float*  k     = (float*)ws;   ws += NE * sizeof(float);
    float*  v     = (float*)ws;   ws += NE * sizeof(float);
    float*  part  = (float*)ws;   // KSPLIT * NE floats

    const dim3 gg(W_ / NT, KSPLIT);
    const int rblk = (int)(NE / 256);

    ln_kernel<<<B_, 256, 0, stream>>>(x, gamma, beta, h_bf);

    gemm_kernel<<<gg, 256, 0, stream>>>(h_bf, Wq, part);
    reduce_kernel<<<rblk, 256, 0, stream>>>(part, bq, nullptr, q);
    gemm_kernel<<<gg, 256, 0, stream>>>(h_bf, Wk, part);
    reduce_kernel<<<rblk, 256, 0, stream>>>(part, bk, nullptr, k);
    gemm_kernel<<<gg, 256, 0, stream>>>(h_bf, Wv, part);
    reduce_kernel<<<rblk, 256, 0, stream>>>(part, bv, nullptr, v);

    softmin_kernel<<<(B_ * 8), 256, 0, stream>>>(q, k, v, h2_bf);

    gemm_kernel<<<gg, 256, 0, stream>>>(h2_bf, Wp, part);
    reduce_kernel<<<rblk, 256, 0, stream>>>(part, bp, x, out);
}
